// GraphSAGE_24094766531338
// MI455X (gfx1250) — matrix operand order repaired and checked by reading, not compiled, not run on hardware
//
#include <hip/hip_runtime.h>
#include <hip/hip_bf16.h>

#define N_NODES 50000
#define N_EDGES 800000
#define DDIM    128
#define M_PAD   50048   // 391 * 128, multiple of (8 waves * 16 rows)

typedef __bf16 bf16;
typedef __attribute__((ext_vector_type(4)))  __bf16 v4bf;
typedef __attribute__((ext_vector_type(8)))  __bf16 v8bf;
typedef __attribute__((ext_vector_type(16))) __bf16 v16bf;
typedef __attribute__((ext_vector_type(8)))  float  v8f;

__device__ __forceinline__ bf16 f2bf(float f) {
    unsigned u = __builtin_bit_cast(unsigned, f);
    u += 0x7FFFu + ((u >> 16) & 1u);           // round to nearest even
    unsigned short h = (unsigned short)(u >> 16);
    return __builtin_bit_cast(bf16, h);
}

// ---------------------------------------------------------------- utilities
__global__ void sage_zero(float* __restrict__ p, int n) {
    for (int i = blockIdx.x * blockDim.x + threadIdx.x; i < n;
         i += gridDim.x * blockDim.x)
        p[i] = 0.0f;
}

__global__ void sage_degree(const int* __restrict__ dst,
                            float* __restrict__ deg, int E) {
    int t = blockIdx.x * blockDim.x + threadIdx.x;
    if (t < E) atomicAdd(&deg[dst[t]], 1.0f);
}

// One edge handled by 32 threads (4 floats each). Edge index is wave-uniform:
// force it into an SGPR so src/dst index loads become scalar loads.
__global__ void sage_scatter(const float* __restrict__ h,
                             const int* __restrict__ src,
                             const int* __restrict__ dst,
                             float* __restrict__ agg, int E) {
    int t = blockIdx.x * blockDim.x + threadIdx.x;
    int e = __builtin_amdgcn_readfirstlane(t >> 5);
    if (e >= E) return;                       // uniform branch
    int c = (threadIdx.x & 31) * 4;
    int s = src[e];
    int d = dst[e];
    const float4 v = *(const float4*)(h + (size_t)s * DDIM + c);
    float* a = agg + (size_t)d * DDIM + c;
    atomicAdd(a + 0, v.x);
    atomicAdd(a + 1, v.y);
    atomicAdd(a + 2, v.z);
    atomicAdd(a + 3, v.w);
}

// mean = agg / max(deg,1); emit bf16 copies of h and mean for the WMMA GEMM,
// and clear agg in the same pass (ready for the next layer's scatter).
__global__ void sage_meanconv(const float* __restrict__ h,
                              float* __restrict__ agg,
                              const float* __restrict__ deg,
                              bf16* __restrict__ hbf,
                              bf16* __restrict__ meanbf) {
    int t = blockIdx.x * blockDim.x + threadIdx.x;   // M_PAD*32 threads
    int row = t >> 5;
    int c = (t & 31) * 4;
    size_t off = (size_t)row * DDIM + c;
    v4bf hb, mb;
    if (row < N_NODES) {
        float inv = 1.0f / fmaxf(deg[row], 1.0f);
        float4 a = *(const float4*)(agg + off);
        float4 hv = *(const float4*)(h + off);
        mb[0] = f2bf(a.x * inv); mb[1] = f2bf(a.y * inv);
        mb[2] = f2bf(a.z * inv); mb[3] = f2bf(a.w * inv);
        hb[0] = f2bf(hv.x); hb[1] = f2bf(hv.y);
        hb[2] = f2bf(hv.z); hb[3] = f2bf(hv.w);
        *(float4*)(agg + off) = make_float4(0.f, 0.f, 0.f, 0.f);
    } else {
        bf16 z = f2bf(0.0f);
        hb[0] = hb[1] = hb[2] = hb[3] = z;
        mb[0] = mb[1] = mb[2] = mb[3] = z;
    }
    *(v4bf*)(hbf + off)    = hb;
    *(v4bf*)(meanbf + off) = mb;
}

// Re-tile W (f32, [128 x nout], row-major) into WMMA-B fragment-major bf16:
// fragment f = kt*(nout/16)+nt holds a 32x16 (KxN) tile; lane l (0..15) covers
// column nt*16+l with K=kt*32+{0..7,16..23}, lane 16..31 the same columns with
// K=kt*32+{8..15,24..31}; each lane's 16 values are contiguous (two b128 loads).
__global__ void sage_prepw(const float* __restrict__ W,
                           bf16* __restrict__ frag, int nout) {
    int t = blockIdx.x * blockDim.x + threadIdx.x;
    int total = DDIM * nout;
    if (t >= total) return;
    int f = t >> 9;          // fragment index (512 elements each)
    int r = t & 511;
    int lane = r >> 4;
    int i = r & 15;
    int nt_count = nout >> 4;
    int kt = f / nt_count;
    int nt = f - kt * nt_count;
    int k = kt * 32 + ((i < 8) ? (lane >> 4) * 8 + i : 16 + (lane >> 4) * 8 + (i - 8));
    int n = nt * 16 + (lane & 15);
    frag[t] = f2bf(W[(size_t)k * nout + n]);
}

// ------------------------------------------------------------------- GEMM
// out[M_PAD, NOUT] = hbf @ Wself + meanbf @ Wneigh + bias  (+ optional SELU)
// 256 threads = 8 waves; each wave owns a 16-row strip. Weights are staged
// into LDS with the CDNA5 async global->LDS DMA path (ASYNCcnt-tracked).
template <int NOUT, int ACT>
__global__ void __launch_bounds__(256) sage_gemm(
    const bf16* __restrict__ hbf, const bf16* __restrict__ meanbf,
    const bf16* __restrict__ wself, const bf16* __restrict__ wneigh,
    const float* __restrict__ bias, float* __restrict__ out) {
    constexpr int KT = DDIM / 32;          // 4 k-tiles
    constexpr int NT = NOUT / 16;
    constexpr int FRAG = 512;              // bf16 elems per 32x16 fragment
    __shared__ __align__(16) bf16 smem[2 * KT * NT * FRAG];

    {   // async global->LDS stage of both weight operands (16B per lane/op).
        // smem is the only LDS object, so its LDS byte offset is 0.
        constexpr int CH = (KT * NT * FRAG) / 8;   // 16B chunks per matrix
        const char* g0 = (const char*)wself;
        const char* g1 = (const char*)wneigh;
#pragma unroll 1
        for (int i = threadIdx.x; i < CH; i += 256) {
            unsigned lds0 = (unsigned)(i * 16);
            unsigned lds1 = (unsigned)((CH + i) * 16);
            unsigned long long a0 = (unsigned long long)(g0 + (size_t)i * 16);
            unsigned long long a1 = (unsigned long long)(g1 + (size_t)i * 16);
            asm volatile("global_load_async_to_lds_b128 %0, %1, off"
                         :: "v"(lds0), "v"(a0) : "memory");
            asm volatile("global_load_async_to_lds_b128 %0, %1, off"
                         :: "v"(lds1), "v"(a1) : "memory");
        }
        asm volatile("s_wait_asynccnt 0x0" ::: "memory");
    }
    __syncthreads();

    const int wave  = threadIdx.x >> 5;
    const int lane  = threadIdx.x & 31;
    const int hlane = lane & 15;
    const int row0  = (blockIdx.x * 8 + wave) * 16;
    const int kb    = (lane < 16) ? 0 : 8;   // A-fragment K sub-offset

    const bf16* ar1 = hbf    + (size_t)(row0 + hlane) * DDIM;
    const bf16* ar2 = meanbf + (size_t)(row0 + hlane) * DDIM;

    // Hoist all A fragments (self + neighbor) into registers.
    v16bf a1[KT], a2[KT];
#pragma unroll
    for (int kt = 0; kt < KT; ++kt) {
        v8bf lo1 = *(const v8bf*)(ar1 + kt * 32 + kb);
        v8bf hi1 = *(const v8bf*)(ar1 + kt * 32 + kb + 16);
        a1[kt] = __builtin_shufflevector(lo1, hi1, 0, 1, 2, 3, 4, 5, 6, 7, 8,
                                         9, 10, 11, 12, 13, 14, 15);
        v8bf lo2 = *(const v8bf*)(ar2 + kt * 32 + kb);
        v8bf hi2 = *(const v8bf*)(ar2 + kt * 32 + kb + 16);
        a2[kt] = __builtin_shufflevector(lo2, hi2, 0, 1, 2, 3, 4, 5, 6, 7, 8,
                                         9, 10, 11, 12, 13, 14, 15);
    }

#pragma unroll
    for (int nt = 0; nt < NT; ++nt) {
        float bval = bias[nt * 16 + hlane];
        v8f acc;
#pragma unroll
        for (int j = 0; j < 8; ++j) acc[j] = bval;

        const bf16* bs = smem + (size_t)nt * FRAG + lane * 16;
        const bf16* bn = bs + KT * NT * FRAG;
#pragma unroll
        for (int kt = 0; kt < KT; ++kt) {
            v16bf b = *(const v16bf*)(bs + (size_t)kt * NT * FRAG);
            acc = __builtin_amdgcn_wmma_f32_16x16x32_bf16(
                false, a1[kt], false, b, (short)0, acc, false, false);
        }
#pragma unroll
        for (int kt = 0; kt < KT; ++kt) {
            v16bf b = *(const v16bf*)(bn + (size_t)kt * NT * FRAG);
            acc = __builtin_amdgcn_wmma_f32_16x16x32_bf16(
                false, a2[kt], false, b, (short)0, acc, false, false);
        }

#pragma unroll
        for (int j = 0; j < 8; ++j) {
            int r = row0 + j + kb;      // C layout: vgpr j -> row j (+8 hi half)
            float v = acc[j];
            if (ACT) {
                // Branchless SELU: kS*max(v,0) + kS*kA*(exp(min(v,0)) - 1)
                const float kS = 1.0507009873554805f;
                const float kA = 1.6732632423543772f;
                float p = fmaxf(v, 0.0f);
                float n = fminf(v, 0.0f);
                v = kS * p + (kS * kA) * (__expf(n) - 1.0f);
            }
            out[(size_t)r * NOUT + nt * 16 + hlane] = v;
        }
    }
}

// ----------------------------------------------------------------- softmax
// One wave32 per row of 64 logits (2 per lane), shuffle reductions.
__global__ void sage_softmax(const float* __restrict__ logits,
                             float* __restrict__ out) {
    int wave = threadIdx.x >> 5;
    int lane = threadIdx.x & 31;
    int row = blockIdx.x * 8 + wave;
    if (row >= N_NODES) return;
    const float* lp = logits + (size_t)row * 64;
    float v0 = lp[lane], v1 = lp[lane + 32];
    float m = fmaxf(v0, v1);
    for (int off = 16; off; off >>= 1) m = fmaxf(m, __shfl_xor(m, off, 32));
    float e0 = __expf(v0 - m), e1 = __expf(v1 - m);
    float s = e0 + e1;
    for (int off = 16; off; off >>= 1) s += __shfl_xor(s, off, 32);
    float inv = 1.0f / s;
    out[(size_t)row * 64 + lane]      = e0 * inv;
    out[(size_t)row * 64 + lane + 32] = e1 * inv;
}

// ------------------------------------------------------------------ launch
extern "C" void kernel_launch(void* const* d_in, const int* in_sizes, int n_in,
                              void* d_out, int out_size, void* d_ws,
                              size_t ws_size, hipStream_t stream) {
    const float* x   = (const float*)d_in[0];
    const int*   src = (const int*)d_in[1];
    const int*   dst = (const int*)d_in[2];
    const float* Ws[3] = {(const float*)d_in[3], (const float*)d_in[6],
                          (const float*)d_in[9]};
    const float* Wn[3] = {(const float*)d_in[4], (const float*)d_in[7],
                          (const float*)d_in[10]};
    const float* B[3]  = {(const float*)d_in[5], (const float*)d_in[8],
                          (const float*)d_in[11]};
    float* outp = (float*)d_out;

    char* w = (char*)d_ws;
    auto alloc = [&](size_t bytes) {
        char* p = w;
        w += (bytes + 255) & ~(size_t)255;
        return p;
    };
    float* deg    = (float*)alloc((size_t)M_PAD * 4);        // contiguous with agg
    float* agg    = (float*)alloc((size_t)M_PAD * DDIM * 4);
    float* h1     = (float*)alloc((size_t)M_PAD * DDIM * 4); // also reused as logits
    float* h2     = (float*)alloc((size_t)M_PAD * DDIM * 4);
    bf16*  hbf    = (bf16*)alloc((size_t)M_PAD * DDIM * 2);
    bf16*  meanbf = (bf16*)alloc((size_t)M_PAD * DDIM * 2);
    bf16*  fs[3], *fn[3];
    fs[0] = (bf16*)alloc(DDIM * 128 * 2);
    fn[0] = (bf16*)alloc(DDIM * 128 * 2);
    fs[1] = (bf16*)alloc(DDIM * 128 * 2);
    fn[1] = (bf16*)alloc(DDIM * 128 * 2);
    fs[2] = (bf16*)alloc(DDIM * 64 * 2);
    fn[2] = (bf16*)alloc(DDIM * 64 * 2);
    float* logits = h1;   // h1 is dead by layer 2

    // deg + agg share one contiguous zero pass (deg block is 256B-aligned size)
    sage_zero<<<2048, 256, 0, stream>>>(deg, M_PAD * (DDIM + 1));
    sage_degree<<<(N_EDGES + 255) / 256, 256, 0, stream>>>(dst, deg, N_EDGES);

    // weight re-tiling (tiny)
    for (int l = 0; l < 3; ++l) {
        int nout = (l == 2) ? 64 : 128;
        int total = DDIM * nout;
        sage_prepw<<<(total + 255) / 256, 256, 0, stream>>>(Ws[l], fs[l], nout);
        sage_prepw<<<(total + 255) / 256, 256, 0, stream>>>(Wn[l], fn[l], nout);
    }

    const int scat_grid = (N_EDGES * 32) / 256;        // exact: 100000
    const int mc_grid   = (M_PAD * 32) / 256;          // exact: 6256
    const int gemm_grid = M_PAD / 128;                 // 391

    // ---- layer 0: x -> h1 (SELU)
    sage_scatter<<<scat_grid, 256, 0, stream>>>(x, src, dst, agg, N_EDGES);
    sage_meanconv<<<mc_grid, 256, 0, stream>>>(x, agg, deg, hbf, meanbf);
    sage_gemm<128, 1><<<gemm_grid, 256, 0, stream>>>(hbf, meanbf, fs[0], fn[0],
                                                     B[0], h1);
    // ---- layer 1: h1 -> h2 (SELU)
    sage_scatter<<<scat_grid, 256, 0, stream>>>(h1, src, dst, agg, N_EDGES);
    sage_meanconv<<<mc_grid, 256, 0, stream>>>(h1, agg, deg, hbf, meanbf);
    sage_gemm<128, 1><<<gemm_grid, 256, 0, stream>>>(hbf, meanbf, fs[1], fn[1],
                                                     B[1], h2);
    // ---- layer 2: h2 -> logits (no act), then softmax
    sage_scatter<<<scat_grid, 256, 0, stream>>>(h2, src, dst, agg, N_EDGES);
    sage_meanconv<<<mc_grid, 256, 0, stream>>>(h2, agg, deg, hbf, meanbf);
    sage_gemm<64, 0><<<gemm_grid, 256, 0, stream>>>(hbf, meanbf, fs[2], fn[2],
                                                    B[2], logits);
    sage_softmax<<<(N_NODES + 7) / 8, 256, 0, stream>>>(logits, outp);
}
